// GIN_12936441495645
// MI455X (gfx1250) — matrix operand order repaired and checked by reading, not compiled, not run on hardware
//
#include <hip/hip_runtime.h>

#define N_NODESC 100000
#define N_EDGESC 1600000
#define N_GRAPHSC 64
#define HID 128
#define NCLS 696
#define N_LAYERSC 4
#define NPAD 100096  // 782 * 128
#define LDSW 136     // padded LDS row stride (elements): 272B = 68 dwords -> no bank conflicts

typedef __attribute__((ext_vector_type(16))) __bf16 v16bf;
typedef __attribute__((ext_vector_type(8)))  __bf16 v8bf;
typedef __attribute__((ext_vector_type(2)))  __bf16 v2bf;
typedef __attribute__((ext_vector_type(8)))  float  v8f;

__device__ __forceinline__ __bf16 f2bf(float f) {
  union { float f; unsigned u; } v; v.f = f;
  unsigned r = v.u + 0x7FFFu + ((v.u >> 16) & 1u);   // round-to-nearest-even
  union { unsigned short s; __bf16 b; } o;
  o.s = (unsigned short)(r >> 16);
  return o.b;
}

// Packed fp32->bf16: one v_cvt_pk_bf16_f32 per 2 elements when available.
#if __has_builtin(__builtin_amdgcn_cvt_pk_bf16_f32)
__device__ __forceinline__ v2bf cvt2(float a, float b) {
  auto t = __builtin_amdgcn_cvt_pk_bf16_f32(a, b);
  union { decltype(t) t_; v2bf v; } u;
  u.t_ = t;
  return u.v;
}
#else
__device__ __forceinline__ v2bf cvt2(float a, float b) {
  v2bf r; r[0] = f2bf(a); r[1] = f2bf(b); return r;
}
#endif

// Low 32 bits of a generic pointer to __shared__ = wave-relative LDS byte
// address (flat->LDS aperture translation uses addr[31:0]).
__device__ __forceinline__ unsigned lds_off_u32(const void* p) {
  return (unsigned)(uintptr_t)p;
}

// GLOBAL_LOAD_ASYNC_TO_LDS_B128 (GV mode): copies 16B from global vaddr to LDS at byte offset vdst.
// Tracked by ASYNCcnt; fence with s_wait_asynccnt.
__device__ __forceinline__ void async_copy_b128(unsigned lds_byte_off, const void* gaddr) {
  asm volatile("global_load_async_to_lds_b128 %0, %1, off"
               :: "v"(lds_byte_off), "v"(gaddr)
               : "memory");
}
__device__ __forceinline__ void wait_async0() {
  asm volatile("s_wait_asynccnt 0" ::: "memory");
}

// ---------------------------------------------------------------------------
// Weight conversion: W[l][k][c] (fp32) -> Wt[l][c][k] (bf16), so B-fragments
// for v_wmma_f32_16x16x32_bf16 are contiguous 32B per lane.
// ---------------------------------------------------------------------------
__global__ __launch_bounds__(256) void convert_w_k(
    const float* __restrict__ W1s, const float* __restrict__ W2s,
    __bf16* __restrict__ W1t, __bf16* __restrict__ W2t) {
  int idx = blockIdx.x * 256 + threadIdx.x;
  if (idx >= N_LAYERSC * HID * HID) return;
  int l = idx >> 14;
  int rem = idx & 16383;
  int k = rem >> 7;
  int c = rem & 127;
  W1t[l * 16384 + c * 128 + k] = f2bf(W1s[l * 16384 + k * 128 + c]);
  W2t[l * 16384 + c * 128 + k] = f2bf(W2s[l * 16384 + k * 128 + c]);
}

// ---------------------------------------------------------------------------
// z = h (rows < n), zero-pad rows [n, NPAD)
// ---------------------------------------------------------------------------
__global__ __launch_bounds__(256) void copy_pad_k(
    const float* __restrict__ src, float* __restrict__ dst, int n) {
  int q = blockIdx.x * 256 + threadIdx.x;         // float4 index
  if (q >= NPAD * HID / 4) return;
  int node = q >> 5;                               // 32 float4 per row
  float4 v = make_float4(0.f, 0.f, 0.f, 0.f);
  if (node < n) v = ((const float4*)src)[q];
  ((float4*)dst)[q] = v;
}

// ---------------------------------------------------------------------------
// Edge scatter-add: one wave per edge. Row gather hits L2 (h fits in 192MB L2)
// ---------------------------------------------------------------------------
__global__ __launch_bounds__(256) void scatter_k(
    const float* __restrict__ h, const long long* __restrict__ ei,
    float* __restrict__ aggr, int E) {
  int e = blockIdx.x * 8 + (threadIdx.x >> 5);
  if (e >= E) return;
  int lane = threadIdx.x & 31;
  long long s = ei[e];
  long long d = ei[(size_t)E + e];
  const float4 v = *(const float4*)(h + (size_t)s * HID + lane * 4);
  float* o = aggr + (size_t)d * HID + lane * 4;
  atomicAdd(o + 0, v.x);
  atomicAdd(o + 1, v.y);
  atomicAdd(o + 2, v.z);
  atomicAdd(o + 3, v.w);
}

// ---------------------------------------------------------------------------
// Fused 2-layer MLP: hout = relu(relu(z@W1+b1)@W2+b2)
// Block = 256 threads = 8 waves; block handles 128 node rows; each wave a
// 16-row strip. WMMA bf16 16x16x32, K-loop of 4, 8 column tiles (N=128).
// Weights are staged into LDS via async copies (ASYNCcnt); GEMM1 output
// round-trips through a padded wave-private LDS strip as bf16.
// ---------------------------------------------------------------------------
__global__ __launch_bounds__(256) void mlp_k(
    const float* __restrict__ z, float* __restrict__ hout,
    const __bf16* __restrict__ W1t, const float* __restrict__ b1,
    const __bf16* __restrict__ W2t, const float* __restrict__ b2) {
  __shared__ __bf16 lw1[HID * LDSW];               // 34 KB, padded rows
  __shared__ __bf16 lw2[HID * LDSW];               // 34 KB
  __shared__ __bf16 z1[8 * 16 * LDSW];             // 34 KB

  const int tid  = threadIdx.x;
  const int lane = tid & 31;
  const int wave = tid >> 5;
  const int hi   = lane >> 4;                      // half-wave select
  const int r    = lane & 15;                      // A-fragment row / C col
  const size_t row0 = (size_t)blockIdx.x * 128 + wave * 16;

  // ---- Stage W1t, W2t into LDS asynchronously (2048 x 16B chunks each) ----
  {
    const unsigned base1 = lds_off_u32(lw1);
    const unsigned base2 = lds_off_u32(lw2);
    const char* g1 = (const char*)W1t;
    const char* g2 = (const char*)W2t;
#pragma unroll
    for (int i = 0; i < 8; ++i) {
      const int c = tid + i * 256;                 // chunk index 0..2047
      const int row = c >> 4;                      // 16 chunks (256B) per row
      const int within = (c & 15) * 16;            // byte within row
      const unsigned loff = row * (LDSW * 2) + within;
      const int goff = row * (HID * 2) + within;
      async_copy_b128(base1 + loff, g1 + goff);
      async_copy_b128(base2 + loff, g2 + goff);
    }
  }

  // ---- A fragments for GEMM1 (fp32 global -> bf16 regs), overlaps asyncs ----
  // 16-bit A 16x32 layout (matching B): lane(row=lane&15) holds K in
  // [16*hi, 16*hi+16) contiguous -> vector elems 0..15.
  union AFrag { v16bf v; v2bf h[8]; };
  AFrag A1[4];
  {
    const float* zr = z + (row0 + r) * HID;
#pragma unroll
    for (int ks = 0; ks < 4; ++ks) {
      const int k0 = ks * 32 + 16 * hi;
      const float4 x0 = *(const float4*)(zr + k0);
      const float4 x1 = *(const float4*)(zr + k0 + 4);
      const float4 x2 = *(const float4*)(zr + k0 + 8);
      const float4 x3 = *(const float4*)(zr + k0 + 12);
      A1[ks].h[0] = cvt2(x0.x, x0.y);
      A1[ks].h[1] = cvt2(x0.z, x0.w);
      A1[ks].h[2] = cvt2(x1.x, x1.y);
      A1[ks].h[3] = cvt2(x1.z, x1.w);
      A1[ks].h[4] = cvt2(x2.x, x2.y);
      A1[ks].h[5] = cvt2(x2.z, x2.w);
      A1[ks].h[6] = cvt2(x3.x, x3.y);
      A1[ks].h[7] = cvt2(x3.z, x3.w);
    }
  }

  wait_async0();       // weights resident in this wave's issued copies
  __syncthreads();     // all waves' copies visible block-wide

  // ---- GEMM1 + bias + ReLU -> z1 (bf16, padded LDS) ----
  // B 32x16 bf16 layout: lane(col=lane&15) holds K in [16*hi, 16*hi+16)
  // contiguous -> 32B load from transposed weights in LDS (conflict-free).
#pragma unroll
  for (int n = 0; n < 8; ++n) {
    v8f acc = 0.0f;
#pragma unroll
    for (int ks = 0; ks < 4; ++ks) {
      const v16bf b = *(const v16bf*)(lw1 + (n * 16 + r) * LDSW + ks * 32 + 16 * hi);
      acc = __builtin_amdgcn_wmma_f32_16x16x32_bf16(
          false, A1[ks].v, false, b, (short)0, acc, false, false);
    }
    const float bias = b1[n * 16 + r];
    // C/D layout: vec elem rr -> row (rr + 8*hi), col (lane&15) of tile
    __bf16* dst = &z1[wave * (16 * LDSW) + (8 * hi) * LDSW + n * 16 + r];
#pragma unroll
    for (int rr = 0; rr < 8; rr += 2) {
      const float va = fmaxf(acc[rr] + bias, 0.f);
      const float vb = fmaxf(acc[rr + 1] + bias, 0.f);
      const v2bf p = cvt2(va, vb);
      dst[(rr)     * LDSW] = p[0];
      dst[(rr + 1) * LDSW] = p[1];
    }
  }

  __syncthreads();  // cross-lane LDS dependency (transpose through LDS)

  // ---- A fragments for GEMM2 from LDS (already bf16) ----
  v16bf A2[4];
  {
    const __bf16* zr = z1 + wave * (16 * LDSW) + r * LDSW;
#pragma unroll
    for (int ks = 0; ks < 4; ++ks) {
      const int k0 = ks * 32 + 16 * hi;
      const v8bf lo  = *(const v8bf*)(zr + k0);
      const v8bf hi8 = *(const v8bf*)(zr + k0 + 8);
#pragma unroll
      for (int j = 0; j < 8; ++j) { A2[ks][j] = lo[j]; A2[ks][8 + j] = hi8[j]; }
    }
  }

  // ---- GEMM2 + bias + ReLU -> global ----
#pragma unroll
  for (int n = 0; n < 8; ++n) {
    v8f acc = 0.0f;
#pragma unroll
    for (int ks = 0; ks < 4; ++ks) {
      const v16bf b = *(const v16bf*)(lw2 + (n * 16 + r) * LDSW + ks * 32 + 16 * hi);
      acc = __builtin_amdgcn_wmma_f32_16x16x32_bf16(
          false, A2[ks], false, b, (short)0, acc, false, false);
    }
    const float bias = b2[n * 16 + r];
#pragma unroll
    for (int rr = 0; rr < 8; ++rr) {
      float v = acc[rr] + bias;
      v = v > 0.f ? v : 0.f;
      hout[(row0 + rr + 8 * hi) * HID + n * 16 + r] = v;
    }
  }
}

// ---------------------------------------------------------------------------
// Small tail kernels
// ---------------------------------------------------------------------------
__global__ __launch_bounds__(256) void zero_k(float* p, int n) {
  int i = blockIdx.x * 256 + threadIdx.x;
  if (i < n) p[i] = 0.f;
}

__global__ __launch_bounds__(256) void pool_k(
    const float* __restrict__ h, const long long* __restrict__ batch,
    float* __restrict__ g, int n) {
  int idx = blockIdx.x * 256 + threadIdx.x;
  if (idx >= n * HID) return;
  int node = idx >> 7;
  int f = idx & 127;
  atomicAdd(&g[(int)batch[node] * HID + f], h[(size_t)node * HID + f]);
}

__global__ __launch_bounds__(128) void fc1_k(
    const float* __restrict__ g, const float* __restrict__ w,
    const float* __restrict__ b, float* __restrict__ g2) {
  int row = blockIdx.x;
  int j = threadIdx.x;
  __shared__ float gs[HID];
  gs[j] = g[row * HID + j];
  __syncthreads();
  float s = b[j];
#pragma unroll 8
  for (int k = 0; k < HID; ++k) s += gs[k] * w[k * HID + j];
  g2[row * HID + j] = s > 0.f ? s : 0.f;
}

__global__ __launch_bounds__(256) void fc2_lsm_k(
    const float* __restrict__ g2, const float* __restrict__ w,
    const float* __restrict__ b, float* __restrict__ out) {
  const int row = blockIdx.x;
  const int t = threadIdx.x;
  __shared__ float gs[HID];
  __shared__ float red[256];
  if (t < HID) gs[t] = g2[row * HID + t];
  __syncthreads();

  float lg[3];
  float m = -3.4e38f;
#pragma unroll
  for (int c = 0; c < 3; ++c) {
    int j = t + c * 256;
    float s = -3.4e38f;
    if (j < NCLS) {
      s = b[j];
#pragma unroll 8
      for (int k = 0; k < HID; ++k) s += gs[k] * w[k * NCLS + j];
    }
    lg[c] = s;
    m = fmaxf(m, s);
  }
  red[t] = m; __syncthreads();
  for (int off = 128; off > 0; off >>= 1) {
    if (t < off) red[t] = fmaxf(red[t], red[t + off]);
    __syncthreads();
  }
  const float mx = red[0];
  __syncthreads();

  float sum = 0.f;
#pragma unroll
  for (int c = 0; c < 3; ++c) {
    int j = t + c * 256;
    if (j < NCLS) sum += __expf(lg[c] - mx);
  }
  red[t] = sum; __syncthreads();
  for (int off = 128; off > 0; off >>= 1) {
    if (t < off) red[t] += red[t + off];
    __syncthreads();
  }
  const float lse = __logf(red[0]) + mx;
#pragma unroll
  for (int c = 0; c < 3; ++c) {
    int j = t + c * 256;
    if (j < NCLS) out[row * NCLS + j] = lg[c] - lse;
  }
}

// ---------------------------------------------------------------------------
// Orchestration
// ---------------------------------------------------------------------------
extern "C" void kernel_launch(void* const* d_in, const int* in_sizes, int n_in,
                              void* d_out, int out_size, void* d_ws, size_t ws_size,
                              hipStream_t stream) {
  const float*     x     = (const float*)d_in[0];
  const long long* ei    = (const long long*)d_in[1];
  const long long* batch = (const long long*)d_in[2];
  const float*     W1s   = (const float*)d_in[3];
  const float*     b1s   = (const float*)d_in[4];
  const float*     W2s   = (const float*)d_in[5];
  const float*     b2s   = (const float*)d_in[6];
  const float*     fc1w  = (const float*)d_in[7];
  const float*     fc1b  = (const float*)d_in[8];
  const float*     fc2w  = (const float*)d_in[9];
  const float*     fc2b  = (const float*)d_in[10];
  float* out = (float*)d_out;

  char* ws = (char*)d_ws;
  size_t off = 0;
  float*  P   = (float*)(ws + off);  off += (size_t)NPAD * HID * 4;   // h
  float*  Q   = (float*)(ws + off);  off += (size_t)NPAD * HID * 4;   // z = h + aggr
  __bf16* W1t = (__bf16*)(ws + off); off += (size_t)N_LAYERSC * HID * HID * 2;
  __bf16* W2t = (__bf16*)(ws + off); off += (size_t)N_LAYERSC * HID * HID * 2;
  float*  g   = (float*)(ws + off);  off += (size_t)N_GRAPHSC * HID * 4;
  float*  g2  = (float*)(ws + off);  off += (size_t)N_GRAPHSC * HID * 4;
  (void)ws_size; (void)in_sizes; (void)n_in; (void)out_size;

  convert_w_k<<<(N_LAYERSC * HID * HID + 255) / 256, 256, 0, stream>>>(W1s, W2s, W1t, W2t);

  for (int l = 0; l < N_LAYERSC; ++l) {
    const float* h = (l == 0) ? x : P;
    copy_pad_k<<<(NPAD * HID / 4 + 255) / 256, 256, 0, stream>>>(h, Q, N_NODESC);
    scatter_k<<<(N_EDGESC + 7) / 8, 256, 0, stream>>>(h, ei, Q, N_EDGESC);
    mlp_k<<<NPAD / 128, 256, 0, stream>>>(Q, P,
                                          W1t + (size_t)l * HID * HID, b1s + l * HID,
                                          W2t + (size_t)l * HID * HID, b2s + l * HID);
  }

  zero_k<<<(2 * N_GRAPHSC * HID + 255) / 256, 256, 0, stream>>>(g, 2 * N_GRAPHSC * HID);
  pool_k<<<(N_NODESC * HID + 255) / 256, 256, 0, stream>>>(P, batch, g, N_NODESC);
  fc1_k<<<N_GRAPHSC, HID, 0, stream>>>(g, fc1w, fc1b, g2);
  fc2_lsm_k<<<N_GRAPHSC, 256, 0, stream>>>(g2, fc2w, fc2b, out);
}
